// BahdanauAttention_softmax_72103910965794
// MI455X (gfx1250) — matrix operand order repaired and checked by reading, not compiled, not run on hardware
//
#include <hip/hip_runtime.h>
#include <hip/hip_bf16.h>

#define BS 4096
#define NQ 21
#define NK 7
#define EDIM 512
#define HDIM 512
#define NEG_INF (-1e9f)

typedef __attribute__((ext_vector_type(16))) __bf16 v16bf;
typedef __attribute__((ext_vector_type(8)))  __bf16 v8bf;
typedef __attribute__((ext_vector_type(8)))  float  v8f;

// Native CDNA5 tanh if the toolchain exposes it (v_tanh_f32), else ocml.
__device__ __forceinline__ float fast_tanh(float x) {
#if __has_builtin(__builtin_amdgcn_tanhf)
  return __builtin_amdgcn_tanhf(x);
#elif __has_builtin(__builtin_amdgcn_tanh_f32)
  return __builtin_amdgcn_tanh_f32(x);
#else
  return tanhf(x);
#endif
}

typedef __attribute__((address_space(3))) float lds_float;

// Async byte-copy global -> LDS (16B per lane), tracked by ASYNCcnt.
__device__ __forceinline__ void async_copy_b128(const float* gsrc, float* ldst) {
  unsigned loff = (unsigned)(uintptr_t)(lds_float*)ldst;  // 32-bit LDS offset
  asm volatile("global_load_async_to_lds_b128 %0, %1, off"
               :: "v"(loff), "v"(gsrc)
               : "memory");
}

__device__ __forceinline__ void wait_async_zero() {
#if __has_builtin(__builtin_amdgcn_s_wait_asynccnt)
  __builtin_amdgcn_s_wait_asynccnt(0);
#else
  asm volatile("s_wait_asynccnt 0x0" ::: "memory");
#endif
}

// ---------------------------------------------------------------------------
// Projection GEMM:  C[M,512] = A[M,512] * W[512,512]^T + bias
// A row-major (m,k); W row-major (n,k) -> both operands are K-major, matching
// the contiguous-K fragment order this code uses for A and B (k = 16*half + s).
// Tile: BM=64 rows per block, BN=512 (full N) -> A read from HBM exactly once,
// W (1MB) stays resident in the 192MB L2 across all blocks.
// 8 waves/block, wave w owns N range [w*64, w*64+64): 4x4 = 16 accum tiles.
// fp32 -> bf16 conversion happens in the staging pass (VGPR path needed).
// ---------------------------------------------------------------------------
__global__ __launch_bounds__(256) void proj_gemm_kernel(
    const float* __restrict__ A, const float* __restrict__ W,
    const float* __restrict__ bias, float* __restrict__ C)
{
  __shared__ __bf16 sA[64 * 32];    // 4 KB
  __shared__ __bf16 sB[512 * 32];   // 32 KB

  const int tid  = threadIdx.x;
  const int wid  = tid >> 5;
  const int lane = tid & 31;
  const int r    = lane & 15;       // row within 16-wide fragment half
  const int half = lane >> 4;       // lane group 0/1 (K-chunk selector)
  const int row0  = blockIdx.x * 64;
  const int nbase = wid * 64;

  v8f acc[4][4];
  const v8f vzero = {0.f,0.f,0.f,0.f,0.f,0.f,0.f,0.f};
#pragma unroll
  for (int i = 0; i < 4; ++i)
#pragma unroll
    for (int j = 0; j < 4; ++j) acc[i][j] = vzero;

  const int arow = tid >> 2;        // 0..63
  const int akc  = (tid & 3) * 8;   // 0,8,16,24
  const int wrow = tid * 2;         // 0..510 (2 rows of W per thread)

  for (int kk = 0; kk < 512; kk += 32) {
    // --- stage A tile (64x32 fp32 -> bf16 LDS), 8 elems/thread ---
    {
      const float* ga = A + (size_t)(row0 + arow) * 512 + kk + akc;
      v8bf pk;
#pragma unroll
      for (int i = 0; i < 8; ++i) pk[i] = (__bf16)ga[i];
      *(v8bf*)(sA + arow * 32 + akc) = pk;
    }
    // --- stage W tile (512x32 fp32 -> bf16 LDS), 64 elems/thread ---
#pragma unroll
    for (int rr = 0; rr < 2; ++rr) {
      const float* gw = W + (size_t)(wrow + rr) * 512 + kk;
#pragma unroll
      for (int c = 0; c < 32; c += 8) {
        v8bf pk;
#pragma unroll
        for (int i = 0; i < 8; ++i) pk[i] = (__bf16)gw[c + i];
        *(v8bf*)(sB + (wrow + rr) * 32 + c) = pk;
      }
    }
    __syncthreads();

    // A fragments: lane(r,half) holds row m=r, K = half*16 .. half*16+15
    v16bf afrag[4];
#pragma unroll
    for (int mt = 0; mt < 4; ++mt) {
      const __bf16* p = sA + (mt * 16 + r) * 32;
      v8bf lo = *(const v8bf*)(p + half * 16);
      v8bf hi = *(const v8bf*)(p + half * 16 + 8);
      v16bf f;
#pragma unroll
      for (int i = 0; i < 8; ++i) { f[i] = lo[i]; f[8 + i] = hi[i]; }
      afrag[mt] = f;
    }
    // B fragments: lane(r,half) holds column n=r, K = half*16 .. half*16+15
#pragma unroll
    for (int nt = 0; nt < 4; ++nt) {
      const __bf16* p = sB + (nbase + nt * 16 + r) * 32 + half * 16;
      v16bf bfrag = *(const v16bf*)p;
#pragma unroll
      for (int mt = 0; mt < 4; ++mt) {
        acc[mt][nt] = __builtin_amdgcn_wmma_f32_16x16x32_bf16(
            false, afrag[mt], false, bfrag, (short)0, acc[mt][nt], false, false);
      }
    }
    __syncthreads();
  }

  // Epilogue: C/D layout -> lane(r,half), VGPR j = element (m = half*8+j, n = r)
#pragma unroll
  for (int nt = 0; nt < 4; ++nt) {
    const int n = nbase + nt * 16 + r;
    const float bv = bias[n];
#pragma unroll
    for (int mt = 0; mt < 4; ++mt) {
      const int m0 = row0 + mt * 16 + half * 8;
#pragma unroll
      for (int j = 0; j < 8; ++j)
        C[(size_t)(m0 + j) * 512 + n] = acc[mt][nt][j] + bv;
    }
  }
}

// ---------------------------------------------------------------------------
// Fused scores -> masked softmax -> weighted key sum. One block per batch b.
// k_proj/keys staged to LDS via GLOBAL_LOAD_ASYNC_TO_LDS_B128 (ASYNCcnt path,
// no VGPR round-trip); each wave owns queries q = wid + 8t. Per lane: 16
// h-values -> v_tanh + FMA -> wave32 shuffle reduction per key. The
// (b,21,7,512) tanh tensor never touches memory.
// ---------------------------------------------------------------------------
__global__ __launch_bounds__(256) void attn_kernel(
    const float* __restrict__ qproj, const float* __restrict__ kproj,
    const float* __restrict__ keys,  const float* __restrict__ wv_w,
    const float* __restrict__ wv_b,  const int* __restrict__ mask,
    float* __restrict__ out, float* __restrict__ out_w)
{
  __shared__ float s_kp[NK * HDIM];     // 14 KB
  __shared__ float s_keys[NK * EDIM];   // 14 KB
  __shared__ float s_wv[HDIM];
  __shared__ float s_madd[NK];

  const int b    = blockIdx.x;
  const int tid  = threadIdx.x;
  const int wid  = tid >> 5;
  const int lane = tid & 31;

  // async-stage k_proj and keys: 896 x 16B chunks each
  const float* gk  = kproj + (size_t)b * NK * HDIM;
  const float* gke = keys  + (size_t)b * NK * EDIM;
  for (int i = tid; i < (NK * HDIM) / 4; i += 256) {
    async_copy_b128(gk  + 4 * i, s_kp   + 4 * i);
    async_copy_b128(gke + 4 * i, s_keys + 4 * i);
  }
  for (int i = tid; i < HDIM; i += 256) s_wv[i] = wv_w[i];
  if (tid < NK) s_madd[tid] = (float)mask[b * NK + tid] * NEG_INF;
  wait_async_zero();
  __syncthreads();

  const float wvb = wv_b[0];

  for (int q = wid; q < NQ; q += 8) {
    const float* gq = qproj + ((size_t)b * NQ + q) * HDIM;
    float qv[16];
#pragma unroll
    for (int i = 0; i < 16; ++i) qv[i] = gq[lane + 32 * i];

    float sc[NK];
#pragma unroll
    for (int k = 0; k < NK; ++k) {
      float p = 0.f;
#pragma unroll
      for (int i = 0; i < 16; ++i) {
        const int h = lane + 32 * i;
        p += fast_tanh(qv[i] + s_kp[k * HDIM + h]) * s_wv[h];
      }
#pragma unroll
      for (int off = 16; off > 0; off >>= 1)
        p += __shfl_down(p, off, 32);
      sc[k] = __shfl(p, 0, 32) + wvb + s_madd[k];
    }

    // softmax over the 7 keys (shift-invariant, in registers)
    float mx = sc[0];
#pragma unroll
    for (int k = 1; k < NK; ++k) mx = fmaxf(mx, sc[k]);
    float sum = 0.f;
    float w[NK];
#pragma unroll
    for (int k = 0; k < NK; ++k) { w[k] = __expf(sc[k] - mx); sum += w[k]; }
    const float inv = 1.f / sum;
#pragma unroll
    for (int k = 0; k < NK; ++k) w[k] *= inv;

    if (lane == 0) {
#pragma unroll
      for (int k = 0; k < NK; ++k)
        out_w[((size_t)b * NQ + q) * NK + k] = w[k];
    }

    float* go = out + ((size_t)b * NQ + q) * EDIM;
#pragma unroll
    for (int i = 0; i < 16; ++i) {
      const int e = lane + 32 * i;
      float a = 0.f;
#pragma unroll
      for (int k = 0; k < NK; ++k) a += w[k] * s_keys[k * EDIM + e];
      go[e] = a;
    }
  }
}

extern "C" void kernel_launch(void* const* d_in, const int* in_sizes, int n_in,
                              void* d_out, int out_size, void* d_ws, size_t ws_size,
                              hipStream_t stream) {
  (void)in_sizes; (void)n_in; (void)out_size; (void)ws_size;
  const float* queries = (const float*)d_in[0];   // (BS,NQ,E)
  const float* keys    = (const float*)d_in[1];   // (BS,NK,E)
  const float* wq_w    = (const float*)d_in[2];   // (H,E)
  const float* wq_b    = (const float*)d_in[3];   // (H)
  const float* wk_w    = (const float*)d_in[4];   // (H,E)
  const float* wk_b    = (const float*)d_in[5];   // (H)
  const float* wv_w    = (const float*)d_in[6];   // (1,H)
  const float* wv_b    = (const float*)d_in[7];   // (1)
  const int*   mask    = (const int*)d_in[8];     // (BS,1,NK,1)

  float* qproj = (float*)d_ws;                          // BS*NQ*H
  float* kproj = qproj + (size_t)BS * NQ * HDIM;        // BS*NK*H

  float* out_attn = (float*)d_out;                      // BS*NQ*E
  float* out_w    = out_attn + (size_t)BS * NQ * EDIM;  // BS*NQ*NK

  // q projection: M = 86016 = 64*1344
  proj_gemm_kernel<<<dim3((BS * NQ) / 64), 256, 0, stream>>>(queries, wq_w, wq_b, qproj);
  // k projection: M = 28672 = 64*448
  proj_gemm_kernel<<<dim3((BS * NK) / 64), 256, 0, stream>>>(keys, wk_w, wk_b, kproj);
  // fused score/softmax/output
  attn_kernel<<<dim3(BS), 256, 0, stream>>>(qproj, kproj, keys, wv_w, wv_b, mask,
                                            out_attn, out_w);
}
